// DecoderRNN_77833397338360
// MI455X (gfx1250) — matrix operand order repaired and checked by reading, not compiled, not run on hardware
//
#include <hip/hip_runtime.h>
#include <hip/hip_bf16.h>
#include <math.h>
#include <stdint.h>

// ---------------------------------------------------------------------------
// DecoderRNN policy rollout on gfx1250 (MI455X, wave32, WMMA).
//
// Shapes: B=64, E=H=512, V=32000, T=20.
//   * One-time: weights -> bf16 in workspace (all weights L2-resident, 192MB).
//   * Per step: fused-K LSTM GEMM -> pointwise cell -> vocab GEMM ->
//     gumbel-argmax sampling -> embedding gather.
//   * GEMM: whole A slab (64xK bf16, 64-128KB) staged in LDS ONCE (WGP has
//     320KB) via async global->LDS (ASYNCcnt), then a barrier-free K loop of
//     ds_load_b128 + global_load_b128 + v_wmma_f32_16x16x32_bf16.
// ---------------------------------------------------------------------------

typedef __attribute__((ext_vector_type(16))) __bf16 v16bf;
typedef __attribute__((ext_vector_type(8)))  __bf16 v8bf;
typedef __attribute__((ext_vector_type(8)))  float  v8f;
typedef __attribute__((ext_vector_type(4)))  int    v4i;

#define B_   64
#define E_   512
#define H_   512
#define V_   32000
#define T_   20
#define KCAT 1024   // E + H
#define G4H  2048   // 4*H

#if __has_builtin(__builtin_amdgcn_global_load_async_to_lds_b128)
#define USE_ASYNC_LDS 1
#define AS1 __attribute__((address_space(1)))
#define AS3 __attribute__((address_space(3)))
#else
#define USE_ASYNC_LDS 0
#endif

// ---------------------------------------------------------------------------
// GEMM: C[64 x N] = A[64 x K](bf16, stride lda) * Brows[N x K]^T
// Weight rows ARE the WMMA B columns (lane n reads two 8-K runs of row n, 16 apart)
// so no transpose is required. 256 threads = 8 wave32; wave computes one
// 16-row tile x four 16-col tiles => WG covers a 64x128 C slab.
// K must be a multiple of 32; N a multiple of 128.
// ---------------------------------------------------------------------------
template <int K>
__global__ __launch_bounds__(256)
void gemm64_wmma_bf16(const __bf16* __restrict__ A, int lda,
                      const __bf16* __restrict__ Brows, int ldb,
                      float* __restrict__ C, int N)
{
    __shared__ __bf16 Atile[64][K];    // 64KB (K=512) / 128KB (K=1024)

    const int tid  = threadIdx.x;
    const int wave = tid >> 5;
    const int lane = tid & 31;
    const int half = lane >> 4;        // 0 = lanes 0-15, 1 = lanes 16-31
    const int l15  = lane & 15;

    const int n_wg   = blockIdx.x << 7;                      // 128-wide N slab
    const int m_tile = (wave & 3) << 4;                      // 0,16,32,48
    const int n_base = n_wg + ((wave >> 2) << 6);            // +0 or +64

    // ---- stage the ENTIRE A slab into LDS once (no K-loop barriers) ------
    constexpr int ROWVECS = K / 8;                           // v8bf per row
    for (int i = tid; i < 64 * ROWVECS; i += 256) {
        const int row = i / ROWVECS;
        const int col = (i % ROWVECS) * 8;
        const __bf16* src = A + (size_t)row * lda + col;
#if USE_ASYNC_LDS
        __builtin_amdgcn_global_load_async_to_lds_b128(
            (AS1 v4i*)(uintptr_t)src,
            (AS3 v4i*)(unsigned)(uintptr_t)&Atile[row][col], 0, 0);
#else
        *reinterpret_cast<v8bf*>(&Atile[row][col]) =
            *reinterpret_cast<const v8bf*>(src);
#endif
    }
#if USE_ASYNC_LDS
#if __has_builtin(__builtin_amdgcn_s_wait_asynccnt)
    __builtin_amdgcn_s_wait_asynccnt(0);
#else
    asm volatile("s_wait_asynccnt 0x0" ::: "memory");
#endif
#endif
    __syncthreads();

    // ---- barrier-free K loop --------------------------------------------
    v8f acc[4] = {{}, {}, {}, {}};

    for (int k0 = 0; k0 < K; k0 += 32) {
        // A fragment (16x32): lane m=l15; K split per half-wave
        v8bf a_lo = *reinterpret_cast<const v8bf*>(
                        &Atile[m_tile + l15][k0 + (half << 3)]);
        v8bf a_hi = *reinterpret_cast<const v8bf*>(
                        &Atile[m_tile + l15][k0 + 16 + (half << 3)]);
        v16bf afrag = __builtin_shufflevector(a_lo, a_hi,
                         0,1,2,3,4,5,6,7,8,9,10,11,12,13,14,15);

        #pragma unroll
        for (int j = 0; j < 4; ++j) {
            // B fragment (32x16): lane n, K split per half-wave (matches A)
            const __bf16* bp = Brows +
                (size_t)(n_base + j * 16 + l15) * ldb + k0 + (half << 3);
            v8bf blo = *reinterpret_cast<const v8bf*>(bp);
            v8bf bhi = *reinterpret_cast<const v8bf*>(bp + 16);
            v16bf bfrag = __builtin_shufflevector(blo, bhi,
                             0,1,2,3,4,5,6,7,8,9,10,11,12,13,14,15);
            if (k0 + 32 < K) __builtin_prefetch(bp + 32, 0, 1);
            acc[j] = __builtin_amdgcn_wmma_f32_16x16x32_bf16(
                         false, afrag, false, bfrag, (short)0, acc[j],
                         false, false);
        }
    }

    // ---- epilogue: C layout = lane n (l15), VGPR r -> M = r + 8*half -----
    #pragma unroll
    for (int j = 0; j < 4; ++j) {
        #pragma unroll
        for (int r = 0; r < 8; ++r) {
            const int m = m_tile + r + (half << 3);
            C[(size_t)m * N + n_base + j * 16 + l15] = acc[j][r];
        }
    }
}

// ---------------------------------------------------------------------------
// One-time conversions
// ---------------------------------------------------------------------------
__global__ __launch_bounds__(256)
void cvt_wcat_kernel(const float* __restrict__ W_ih, const float* __restrict__ W_hh,
                     __bf16* __restrict__ Wcat)
{
    int idx = blockIdx.x * 256 + threadIdx.x;          // 2048*1024
    if (idx >= G4H * KCAT) return;
    int n = idx >> 10, k = idx & (KCAT - 1);
    float v = (k < E_) ? W_ih[n * E_ + k] : W_hh[n * H_ + (k - E_)];
    Wcat[idx] = (__bf16)v;
}

__global__ __launch_bounds__(256)
void cvt_linw_kernel(const float* __restrict__ linear_W, __bf16* __restrict__ linW)
{
    size_t idx = (size_t)blockIdx.x * 256 + threadIdx.x; // 32000*512
    if (idx >= (size_t)V_ * H_) return;
    linW[idx] = (__bf16)linear_W[idx];
}

__global__ __launch_bounds__(256)
void init_xcat_kernel(const float* __restrict__ features, __bf16* __restrict__ xcat)
{
    int idx = blockIdx.x * 256 + threadIdx.x;           // 64*1024
    if (idx >= B_ * KCAT) return;
    int b = idx >> 10, e = idx & (KCAT - 1);
    xcat[idx] = (e < E_) ? (__bf16)features[b * E_ + e] : (__bf16)0.0f;
}

// ---------------------------------------------------------------------------
// LSTM pointwise cell: gates[64 x 2048] (i,f,g,o blocks) -> c, h
// ---------------------------------------------------------------------------
__device__ __forceinline__ float sigmoid_f(float x) {
    return 1.0f / (1.0f + __expf(-x));
}
__device__ __forceinline__ float tanh_f(float x) {
    float e = __expf(2.0f * x);
    return (e - 1.0f) / (e + 1.0f);
}

__global__ __launch_bounds__(256)
void lstm_cell_kernel(const float* __restrict__ gates,
                      const float* __restrict__ b_ih, const float* __restrict__ b_hh,
                      float* __restrict__ cbuf, __bf16* __restrict__ xcat, int t)
{
    int idx = blockIdx.x * 256 + threadIdx.x;          // 64*512
    if (idx >= B_ * H_) return;
    int b = idx >> 9, j = idx & (H_ - 1);
    const float* g = gates + (size_t)b * G4H;
    float gi = g[j]          + b_ih[j]          + b_hh[j];
    float gf = g[H_ + j]     + b_ih[H_ + j]     + b_hh[H_ + j];
    float gg = g[2 * H_ + j] + b_ih[2 * H_ + j] + b_hh[2 * H_ + j];
    float go = g[3 * H_ + j] + b_ih[3 * H_ + j] + b_hh[3 * H_ + j];
    float i = sigmoid_f(gi), f = sigmoid_f(gf), o = sigmoid_f(go), gv = tanh_f(gg);
    float cprev = (t == 0) ? 0.0f : cbuf[idx];
    float cnew  = f * cprev + i * gv;
    float h     = o * tanh_f(cnew);
    cbuf[idx] = cnew;
    xcat[(size_t)b * KCAT + E_ + j] = (__bf16)h;
}

// ---------------------------------------------------------------------------
// Per-row sampling: log-softmax + gumbel-argmax categorical, EOS masking.
// ---------------------------------------------------------------------------
__device__ __forceinline__ float u01_hash(unsigned t, unsigned b, unsigned v)
{
    unsigned long long x = ((unsigned long long)t << 44) ^
                           ((unsigned long long)b << 36) ^
                           ((unsigned long long)v)       ^ 0x9E3779B97F4A7C15ull;
    x ^= x >> 33; x *= 0xff51afd7ed558ccdull;
    x ^= x >> 33; x *= 0xc4ceb9fe1a85ec53ull;
    x ^= x >> 33;
    return (float)((x >> 40) + 1ull) * (1.0f / 16777218.0f);   // (0,1)
}

__global__ __launch_bounds__(256)
void sample_kernel(const float* __restrict__ logits, const float* __restrict__ linear_b,
                   float* __restrict__ out, int* __restrict__ tokbuf,
                   int* __restrict__ done, int t)
{
    __shared__ float sred[256];
    __shared__ int   sidx[256];
    const int b   = blockIdx.x;
    const int tid = threadIdx.x;
    const float* row = logits + (size_t)b * V_;

    // pass 1: row max
    float m = -3.4e38f;
    for (int v = tid; v < V_; v += 256) m = fmaxf(m, row[v] + linear_b[v]);
    sred[tid] = m; __syncthreads();
    for (int s = 128; s > 0; s >>= 1) {
        if (tid < s) sred[tid] = fmaxf(sred[tid], sred[tid + s]);
        __syncthreads();
    }
    const float rmax = sred[0]; __syncthreads();

    // pass 2: log-sum-exp
    float sum = 0.0f;
    for (int v = tid; v < V_; v += 256) sum += __expf(row[v] + linear_b[v] - rmax);
    sred[tid] = sum; __syncthreads();
    for (int s = 128; s > 0; s >>= 1) {
        if (tid < s) sred[tid] += sred[tid + s];
        __syncthreads();
    }
    const float logZ = rmax + __logf(sred[0]); __syncthreads();

    // pass 3: gumbel-argmax sample
    float best = -3.4e38f; int bi = 0;
    for (int v = tid; v < V_; v += 256) {
        float u   = u01_hash((unsigned)t, (unsigned)b, (unsigned)v);
        float key = row[v] + linear_b[v] - __logf(-__logf(u));
        if (key > best) { best = key; bi = v; }
    }
    sred[tid] = best; sidx[tid] = bi; __syncthreads();
    for (int s = 128; s > 0; s >>= 1) {
        if (tid < s && sred[tid + s] > sred[tid]) {
            sred[tid] = sred[tid + s]; sidx[tid] = sidx[tid + s];
        }
        __syncthreads();
    }

    if (tid == 0) {
        const int   tk    = sidx[0];
        const float lp    = row[tk] + linear_b[tk] - logZ;
        const int   dprev = (t == 0) ? 0 : done[b];
        out[b * T_ + t]           = dprev ? 0.0f : (float)tk;   // sampled id
        out[B_ * T_ + b * T_ + t] = dprev ? 0.0f : lp;          // log prob
        done[b]   = dprev | (tk == 2);                          // EOS=2
        tokbuf[b] = tk;                                         // feeds embed
    }
}

__global__ __launch_bounds__(256)
void embed_kernel(const float* __restrict__ embed_W, const int* __restrict__ tokbuf,
                  __bf16* __restrict__ xcat)
{
    int idx = blockIdx.x * 256 + threadIdx.x;          // 64*512
    if (idx >= B_ * E_) return;
    int b = idx >> 9, e = idx & (E_ - 1);
    xcat[(size_t)b * KCAT + e] = (__bf16)embed_W[(size_t)tokbuf[b] * E_ + e];
}

// ---------------------------------------------------------------------------
// Host-side orchestration (graph-capturable: only kernel launches on stream)
// ---------------------------------------------------------------------------
extern "C" void kernel_launch(void* const* d_in, const int* in_sizes, int n_in,
                              void* d_out, int out_size, void* d_ws, size_t ws_size,
                              hipStream_t stream)
{
    (void)in_sizes; (void)n_in; (void)out_size; (void)ws_size;

    const float* features = (const float*)d_in[0];
    // d_in[1] captions, d_in[2] lengths, d_in[3] use_policy: unused here
    const float* embed_W  = (const float*)d_in[4];
    const float* W_ih     = (const float*)d_in[5];
    const float* W_hh     = (const float*)d_in[6];
    const float* b_ih     = (const float*)d_in[7];
    const float* b_hh     = (const float*)d_in[8];
    const float* linear_W = (const float*)d_in[9];
    const float* linear_b = (const float*)d_in[10];
    float* out = (float*)d_out;

    // ---- workspace carve-up ----
    char* ws = (char*)d_ws;
    size_t off = 0;
    auto carve = [&](size_t bytes) {
        char* p = ws + off;
        off += (bytes + 255) & ~(size_t)255;
        return p;
    };
    __bf16* Wcat   = (__bf16*)carve((size_t)G4H * KCAT * sizeof(__bf16)); // 4 MB
    __bf16* linW   = (__bf16*)carve((size_t)V_ * H_ * sizeof(__bf16));    // 32 MB
    __bf16* xcat   = (__bf16*)carve((size_t)B_ * KCAT * sizeof(__bf16));  // 128 KB
    float*  gates  = (float*) carve((size_t)B_ * G4H * sizeof(float));    // 512 KB
    float*  cbuf   = (float*) carve((size_t)B_ * H_ * sizeof(float));     // 128 KB
    float*  logits = (float*) carve((size_t)B_ * V_ * sizeof(float));     // 8 MB
    int*    tokbuf = (int*)   carve(B_ * sizeof(int));
    int*    done   = (int*)   carve(B_ * sizeof(int));

    // ---- one-time weight conversion + initial x = features, h = 0 ----
    cvt_wcat_kernel<<<(G4H * KCAT + 255) / 256, 256, 0, stream>>>(W_ih, W_hh, Wcat);
    cvt_linw_kernel<<<(int)(((size_t)V_ * H_ + 255) / 256), 256, 0, stream>>>(linear_W, linW);
    init_xcat_kernel<<<(B_ * KCAT + 255) / 256, 256, 0, stream>>>(features, xcat);

    // ---- autoregressive rollout ----
    for (int t = 0; t < T_; ++t) {
        // gates = [x|h] @ [W_ih|W_hh]^T   (K=1024, N=2048, 16 WGs)
        gemm64_wmma_bf16<KCAT><<<G4H / 128, 256, 0, stream>>>(
            xcat, KCAT, Wcat, KCAT, gates, G4H);
        // LSTM cell pointwise; writes h (bf16) into xcat[:, 512:]
        lstm_cell_kernel<<<(B_ * H_ + 255) / 256, 256, 0, stream>>>(
            gates, b_ih, b_hh, cbuf, xcat, t);
        // logits = h @ linear_W^T          (K=512, N=32000, 250 WGs)
        gemm64_wmma_bf16<H_><<<V_ / 128, 256, 0, stream>>>(
            xcat + E_, KCAT, linW, H_, logits, V_);
        // log-softmax + categorical sample + EOS bookkeeping
        sample_kernel<<<B_, 256, 0, stream>>>(
            logits, linear_b, out, tokbuf, done, t);
        // next-step input embedding (not needed after the last step)
        if (t + 1 < T_) {
            embed_kernel<<<(B_ * E_ + 255) / 256, 256, 0, stream>>>(
                embed_W, tokbuf, xcat);
        }
    }
}
